// HGCNEncoder_34909494182550
// MI455X (gfx1250) — compile-verified, matrix operand order checked by reading, not run on hardware
//
#include <hip/hip_runtime.h>
#include <hip/hip_bf16.h>

#define DIM 256
#define LDSP 264            // padded LDS row stride (bf16 elems): 528B -> conflict-free
#define EPSV 1e-5f

typedef __attribute__((ext_vector_type(16))) __bf16 v16bf;
typedef __attribute__((ext_vector_type(8)))  float  v8f;

// ---------- helpers ----------
__device__ __forceinline__ unsigned short f2bf(float f) {
    unsigned int u = __float_as_uint(f);
    unsigned int r = u + 0x7FFFu + ((u >> 16) & 1u);   // round-to-nearest-even
    return (unsigned short)(r >> 16);
}

__device__ __forceinline__ v16bf load_frag(const unsigned short* p0,
                                           const unsigned short* p1) {
    union { v16bf v; uint4 q[2]; } u;
    u.q[0] = *(const uint4*)p0;   // 8 bf16, K = kh..kh+7
    u.q[1] = *(const uint4*)p1;   // 8 bf16, K = kh+16..kh+23
    return u.v;
}

// ---------- elementwise / utility kernels ----------
__global__ __launch_bounds__(256) void k_zero_f32(float* __restrict__ p, int n) {
    int i = blockIdx.x * 256 + threadIdx.x;
    if (i < n) p[i] = 0.0f;
}

__global__ __launch_bounds__(256) void k_f32_to_bf16(const float* __restrict__ s,
                                                     unsigned short* __restrict__ d, int n) {
    int i = blockIdx.x * 256 + threadIdx.x;
    if (i < n) d[i] = f2bf(s[i]);
}

// wb[n][k] = bf16(w[k][n])  (transpose 256x256, weights stored [fan_in][fan_out])
__global__ __launch_bounds__(256) void k_wtrans(const float* __restrict__ w,
                                                unsigned short* __restrict__ wb) {
    int n = blockIdx.x;          // output column (N)
    int k = threadIdx.x;         // input row (K)
    wb[n * DIM + k] = f2bf(w[k * DIM + n]);
}

__global__ __launch_bounds__(256) void k_degree(const int* __restrict__ ni,
                                                const int* __restrict__ ei,
                                                float* __restrict__ D,
                                                float* __restrict__ B, int nnz) {
    int i = blockIdx.x * 256 + threadIdx.x;
    if (i < nnz) {
        atomicAdd(&D[ni[i]], 1.0f);
        atomicAdd(&B[ei[i]], 1.0f);
    }
}

__global__ __launch_bounds__(256) void k_invert(float* __restrict__ p, int n) {
    int i = blockIdx.x * 256 + threadIdx.x;
    if (i < n) { float v = p[i]; p[i] = (v > 0.0f) ? 1.0f / v : 0.0f; }
}

// dst[sidx[i]] += src[gidx[i]]  (row of DIM floats per nnz; 64 lanes x float4)
__global__ __launch_bounds__(256) void k_scatter_add(const float* __restrict__ src,
                                                     float* __restrict__ dst,
                                                     const int* __restrict__ gidx,
                                                     const int* __restrict__ sidx, int nnz) {
    int i = blockIdx.x * 4 + (threadIdx.x >> 6);
    if (i >= nnz) return;
    int c = (threadIdx.x & 63) * 4;
    const float4 v = *(const float4*)(src + (size_t)gidx[i] * DIM + c);
    float* d = dst + (size_t)sidx[i] * DIM + c;
    atomicAdd(d + 0, v.x); atomicAdd(d + 1, v.y);
    atomicAdd(d + 2, v.z); atomicAdd(d + 3, v.w);
}

__global__ __launch_bounds__(256) void k_scale_rows(float* __restrict__ e,
                                                    const float* __restrict__ binv, int n) {
    int i = blockIdx.x * 256 + threadIdx.x;
    if (i < n) e[i] *= binv[i >> 8];
}

// bf_out = bf16(y * Dinv[row] + bias[col])
__global__ __launch_bounds__(256) void k_scale_bias_bf16(const float* __restrict__ y,
                                                         const float* __restrict__ dinv,
                                                         const float* __restrict__ bias,
                                                         unsigned short* __restrict__ o, int n) {
    int i = blockIdx.x * 256 + threadIdx.x;
    if (i < n) o[i] = f2bf(y[i] * dinv[i >> 8] + bias[i & (DIM - 1)]);
}

// ---------- WMMA GEMM with async-LDS A staging ----------
// C[MxDIM] = A[MxDIM](bf16) * W(bf16, [N][K]) (+bias)(relu?)
// Output: Cf (f32) if Cbf==nullptr, else Cbf (bf16). In-place A==Cbf is safe:
// the whole 16-row A strip is staged into LDS before any store.
__global__ __launch_bounds__(256) void k_gemm_wmma(const unsigned short* __restrict__ A,
                                                   const unsigned short* __restrict__ BT,
                                                   const float* __restrict__ bias,
                                                   float* __restrict__ Cf,
                                                   unsigned short* __restrict__ Cbf,
                                                   int relu) {
    __shared__ __attribute__((aligned(16))) unsigned short As[16 * LDSP];

    const int t    = threadIdx.x;
    const int row0 = blockIdx.x * 16;

    // Stage 16x256 bf16 A strip (8 KB) into LDS: 512 16B chunks, 2 per thread,
    // using the gfx1250 async copy path (tracked by ASYNCcnt).
    #pragma unroll
    for (int j = 0; j < 2; ++j) {
        int chunk = t + j * 256;             // 0..511
        int r = chunk >> 5;                  // 16 rows
        int c = (chunk & 31) * 8;            // 32 chunks of 8 bf16 per row
        const unsigned short* gp = A + (size_t)(row0 + r) * DIM + c;
        unsigned lp = (unsigned)(uintptr_t)(&As[r * LDSP + c]);
        asm volatile("global_load_async_to_lds_b128 %0, %1, off"
                     :: "v"(lp), "v"(gp) : "memory");
    }
    asm volatile("s_wait_asynccnt 0x0" ::: "memory");
    __syncthreads();

    const int lane = t & 31;
    const int wave = t >> 5;                 // 0..7
    const int m    = lane & 15;              // row/col within 16-tile
    const int kh   = (lane >> 4) * 8;        // K-half offset per ISA A/B layout

    const unsigned short* arow = As + m * LDSP;
    const int n0 = wave * 2, n1 = n0 + 1;    // two 16-wide N tiles per wave
    const unsigned short* b0r = BT + (size_t)(n0 * 16 + m) * DIM;
    const unsigned short* b1r = BT + (size_t)(n1 * 16 + m) * DIM;

    v8f c0 = {}; v8f c1 = {};
    #pragma unroll
    for (int kk = 0; kk < DIM; kk += 32) {
        v16bf a  = load_frag(arow + kk + kh, arow + kk + kh + 16);   // ds_load_b128 x2
        v16bf b0 = load_frag(b0r  + kk + kh, b0r  + kk + kh + 16);
        v16bf b1 = load_frag(b1r  + kk + kh, b1r  + kk + kh + 16);
        c0 = __builtin_amdgcn_wmma_f32_16x16x32_bf16(false, a, false, b0,
                                                     (short)0, c0, false, false);
        c1 = __builtin_amdgcn_wmma_f32_16x16x32_bf16(false, a, false, b1,
                                                     (short)0, c1, false, false);
    }

    const int nc0 = n0 * 16 + m, nc1 = n1 * 16 + m;
    const float bb0 = bias ? bias[nc0] : 0.0f;
    const float bb1 = bias ? bias[nc1] : 0.0f;
    #pragma unroll
    for (int e = 0; e < 8; ++e) {
        int r = row0 + kh + e;               // D layout: M = e + 8*(lane>>4)
        float v0 = c0[e] + bb0;
        float v1 = c1[e] + bb1;
        if (relu) { v0 = fmaxf(v0, 0.0f); v1 = fmaxf(v1, 0.0f); }
        if (Cbf) {
            Cbf[(size_t)r * DIM + nc0] = f2bf(v0);
            Cbf[(size_t)r * DIM + nc1] = f2bf(v1);
        } else {
            Cf[(size_t)r * DIM + nc0] = v0;
            Cf[(size_t)r * DIM + nc1] = v1;
        }
    }
}

// ---------- BatchNorm ----------
__global__ __launch_bounds__(256) void k_bn_stats(const float* __restrict__ z,
                                                  float* __restrict__ sum,
                                                  float* __restrict__ sumsq, int rows) {
    int c = threadIdx.x;
    int r0 = blockIdx.x * 250;
    int r1 = r0 + 250; if (r1 > rows) r1 = rows;
    float s = 0.0f, q = 0.0f;
    for (int r = r0; r < r1; ++r) {
        float v = z[(size_t)r * DIM + c];
        s += v; q += v * v;
    }
    atomicAdd(&sum[c], s);
    atomicAdd(&sumsq[c], q);
}

__global__ __launch_bounds__(256) void k_bn_final(const float* __restrict__ sum,
                                                  const float* __restrict__ sumsq,
                                                  const float* __restrict__ g,
                                                  const float* __restrict__ be,
                                                  float* __restrict__ scale,
                                                  float* __restrict__ shift, int rows) {
    int c = threadIdx.x;
    float inv_n = 1.0f / (float)rows;
    float mu  = sum[c] * inv_n;
    float var = sumsq[c] * inv_n - mu * mu;
    float is  = g[c] * rsqrtf(var + EPSV);
    scale[c] = is;
    shift[c] = be[c] - mu * is;
}

// Writes f32 (of) when ob==nullptr, else bf16 (ob) — next consumer is a bf16 GEMM.
__global__ __launch_bounds__(256) void k_bn_apply(const float* __restrict__ z,
                                                  const float* __restrict__ scale,
                                                  const float* __restrict__ shift,
                                                  float* __restrict__ of,
                                                  unsigned short* __restrict__ ob, int n) {
    int i = blockIdx.x * 256 + threadIdx.x;
    if (i < n) {
        int c = i & (DIM - 1);
        float v = z[i] * scale[c] + shift[c];
        if (ob) ob[i] = f2bf(v);
        else    of[i] = v;
    }
}

// ---------- launch ----------
extern "C" void kernel_launch(void* const* d_in, const int* in_sizes, int n_in,
                              void* d_out, int out_size, void* d_ws, size_t ws_size,
                              hipStream_t stream) {
    const float* x_in = (const float*)d_in[0];
    const int*   eidx = (const int*)d_in[1];
    const int N   = in_sizes[0] / DIM;       // 100000
    const int nnz = in_sizes[1] / 2;         // 3200000
    const int* ni = eidx;                    // node_idx  (row 0)
    const int* ei = eidx + nnz;              // edge_idx  (row 1)
    const int E = 20000;

    // workspace carve-out
    size_t off = 0;
    auto alloc = [&](size_t bytes) -> char* {
        char* p = (char*)d_ws + off;
        off = (off + bytes + 255) & ~(size_t)255;
        return p;
    };
    float*          buf0  = (float*)alloc((size_t)N * DIM * 4);
    float*          buf1  = (float*)alloc((size_t)N * DIM * 4);
    unsigned short* bfbuf = (unsigned short*)alloc((size_t)N * DIM * 2);
    float*          ebuf  = (float*)alloc((size_t)E * DIM * 4);
    unsigned short* wb    = (unsigned short*)alloc((size_t)DIM * DIM * 2);
    float*          Dinv  = (float*)alloc((size_t)N * 4);
    float*          Binv  = (float*)alloc((size_t)E * 4);
    float*          sums  = (float*)alloc(DIM * 4);
    float*          sumsq = (float*)alloc(DIM * 4);
    float*          scale = (float*)alloc(DIM * 4);
    float*          shift = (float*)alloc(DIM * 4);

    const int NE = N * DIM;     // 25.6M
    const int EE = E * DIM;     // 5.12M
    dim3 t256(256);
    auto g1 = [](int n) { return dim3((unsigned)((n + 255) / 256)); };

    // degrees (once per launch; ws may be poisoned)
    k_zero_f32<<<g1(N), t256, 0, stream>>>(Dinv, N);
    k_zero_f32<<<g1(E), t256, 0, stream>>>(Binv, E);
    k_degree<<<g1(nnz), t256, 0, stream>>>(ni, ei, Dinv, Binv, nnz);
    k_invert<<<g1(N), t256, 0, stream>>>(Dinv, N);
    k_invert<<<g1(E), t256, 0, stream>>>(Binv, E);

    dim3 gGemm((unsigned)(N / 16));                     // 6250 M-tiles
    dim3 gScat((unsigned)((nnz + 3) / 4));              // 4 nnz per block
    dim3 gStat((unsigned)((N + 249) / 250));            // 250 rows per block

    // initial activation -> bf16 (later blocks get bf16 straight from BN-apply)
    k_f32_to_bf16<<<g1(NE), t256, 0, stream>>>(x_in, bfbuf, NE);

    for (int l = 0; l < 3; ++l) {
        const float* cw = (const float*)d_in[2 + 8 * l + 0];
        const float* cb = (const float*)d_in[2 + 8 * l + 1];
        const float* aw = (const float*)d_in[2 + 8 * l + 2];
        const float* ab = (const float*)d_in[2 + 8 * l + 3];
        const float* bw = (const float*)d_in[2 + 8 * l + 4];
        const float* bb = (const float*)d_in[2 + 8 * l + 5];
        const float* gg = (const float*)d_in[2 + 8 * l + 6];
        const float* be = (const float*)d_in[2 + 8 * l + 7];

        // 1) xw = X @ cw   (f32 out for the scatter phase)
        k_wtrans<<<dim3(DIM), t256, 0, stream>>>(cw, wb);
        k_gemm_wmma<<<gGemm, t256, 0, stream>>>(bfbuf, wb, nullptr, buf1, nullptr, 0);

        // 2) e = segsum(xw[node] -> edge) * Binv
        k_zero_f32<<<g1(EE), t256, 0, stream>>>(ebuf, EE);
        k_scatter_add<<<gScat, t256, 0, stream>>>(buf1, ebuf, ni, ei, nnz);
        k_scale_rows<<<g1(EE), t256, 0, stream>>>(ebuf, Binv, EE);

        // 3) y = segsum(e[edge] -> node) * Dinv + cb  (fused scale+bias+bf16)
        k_zero_f32<<<g1(NE), t256, 0, stream>>>(buf0, NE);
        k_scatter_add<<<gScat, t256, 0, stream>>>(ebuf, buf0, ei, ni, nnz);
        k_scale_bias_bf16<<<g1(NE), t256, 0, stream>>>(buf0, Dinv, cb, bfbuf, NE);

        // 4) h = relu(y @ aw + ab)  -> bf16 IN PLACE over bfbuf (A staged in LDS)
        k_wtrans<<<dim3(DIM), t256, 0, stream>>>(aw, wb);
        k_gemm_wmma<<<gGemm, t256, 0, stream>>>(bfbuf, wb, ab, nullptr, bfbuf, 1);

        // 5) z = relu(h @ bw + bb)  -> f32 for BN stats
        k_wtrans<<<dim3(DIM), t256, 0, stream>>>(bw, wb);
        k_gemm_wmma<<<gGemm, t256, 0, stream>>>(bfbuf, wb, bb, buf0, nullptr, 1);

        // 6) BatchNorm (training-mode, biased var)
        k_zero_f32<<<dim3(1), t256, 0, stream>>>(sums, DIM);
        k_zero_f32<<<dim3(1), t256, 0, stream>>>(sumsq, DIM);
        k_bn_stats<<<gStat, t256, 0, stream>>>(buf0, sums, sumsq, N);
        k_bn_final<<<dim3(1), t256, 0, stream>>>(sums, sumsq, gg, be, scale, shift, N);
        if (l == 2) {
            k_bn_apply<<<g1(NE), t256, 0, stream>>>(buf0, scale, shift,
                                                    (float*)d_out, nullptr, NE);
        } else {
            k_bn_apply<<<g1(NE), t256, 0, stream>>>(buf0, scale, shift,
                                                    nullptr, bfbuf, NE);
        }
    }
    (void)n_in; (void)out_size; (void)ws_size;
}
